// FFM_79250736546626
// MI455X (gfx1250) — matrix operand order repaired and checked by reading, not compiled, not run on hardware
//
#include <hip/hip_runtime.h>
#include <hip/hip_bf16.h>

typedef __attribute__((ext_vector_type(16))) _Float16 v16h;
typedef __attribute__((ext_vector_type(8)))  float    v8f;
typedef __attribute__((ext_vector_type(4)))  float    v4f;

#define NFIELD 20
#define VFIELD 5000
#define NTOT   100000
#define DIM    64
#define NPAIR  190
#define WPB    8          // waves per block (wave32)
#define CHUNK  ((NPAIR + WPB - 1) / WPB)   // 24 pairs per wave

// Convert 8 contiguous f32 at p0 and 8 contiguous f32 at p1 into a v16h fragment.
__device__ __forceinline__ v16h ldcvt16(const float* p0, const float* p1) {
    v4f x0 = *(const v4f*)(p0);
    v4f x1 = *(const v4f*)(p0 + 4);
    v4f y0 = *(const v4f*)(p1);
    v4f y1 = *(const v4f*)(p1 + 4);
    v16h r;
#pragma unroll
    for (int e = 0; e < 4; ++e) {
        r[e]      = (_Float16)x0[e];
        r[4 + e]  = (_Float16)x1[e];
        r[8 + e]  = (_Float16)y0[e];
        r[12 + e] = (_Float16)y1[e];
    }
    return r;
}

__global__ __launch_bounds__(32 * WPB)
void ffm_forward(const int* __restrict__ x,      // (B, 20) int32
                 const float* __restrict__ W,    // (20, 100000, 64) f32
                 const float* __restrict__ Wl,   // (100000, 1) f32
                 const float* __restrict__ bias, // (1,) f32
                 float* __restrict__ out)        // (B, 1) f32
{
    __shared__ int   sIdx[16 * NFIELD];     // offset-adjusted row indices for tile
    __shared__ float sRed[WPB * 256];       // per-wave partial 16x16 accumulators

    const int tid  = threadIdx.x;
    const int wave = __builtin_amdgcn_readfirstlane(tid >> 5);  // scalar wave id
    const int lane = tid & 31;
    const int srow = lane & 15;             // sample-in-tile this lane serves
    const int s0   = blockIdx.x * 16;       // tile base sample

    // Stage row indices (x + field*5000) into LDS, shared by all waves.
    for (int t = tid; t < 16 * NFIELD; t += blockDim.x) {
        int s = t / NFIELD, f = t % NFIELD;
        sIdx[t] = x[(s0 + s) * NFIELD + f] + f * VFIELD;
    }
    __syncthreads();

    // WMMA fragment k-offsets per ISA 7.12.2:
    //  A (16x32 f16): lanes 0-15 hold K {0..7,16..23}, lanes 16-31 hold K {8..15,24..31}
    //  B (32x16 f16): lanes 0-15 hold K {0..7,16..23} of col n=lane, lanes 16-31 hold K {8..15,24..31}
    const int koffA = (lane < 16) ? 0 : 8;
    const int koffB = (lane < 16) ? 0 : 8;
    const int rowb  = srow * NFIELD;

    // Contiguous pair range for this wave; decode start (i,j) ONCE, then
    // update incrementally inside the loop (no per-iteration decode).
    const int pbeg = wave * CHUNK;
    const int pend = (pbeg + CHUNK < NPAIR) ? (pbeg + CHUNK) : NPAIR;
    int i = 0, pr = pbeg;
    while (pr >= (NFIELD - 1 - i)) { pr -= (NFIELD - 1 - i); ++i; }
    int j = i + 1 + pr;

    v8f c = {};   // f32 accumulator: diagonal collects per-sample interaction sum

    for (int p = pbeg; p < pend; ++p) {
        const int idx_i = sIdx[rowb + i];
        const int idx_j = sIdx[rowb + j];
        const float* pA = W + ((size_t)j * NTOT + (size_t)idx_i) * DIM;
        const float* pB = W + ((size_t)i * NTOT + (size_t)idx_j) * DIM;

        // K = 0..31
        v16h a0 = ldcvt16(pA + koffA,      pA + koffA + 16);
        v16h b0 = ldcvt16(pB + koffB,      pB + koffB + 16);
        c = __builtin_amdgcn_wmma_f32_16x16x32_f16(false, a0, false, b0,
                                                   (short)0, c, false, false);
        // K = 32..63
        v16h a1 = ldcvt16(pA + 32 + koffA, pA + 32 + koffA + 16);
        v16h b1 = ldcvt16(pB + 32 + koffB, pB + 32 + koffB + 16);
        c = __builtin_amdgcn_wmma_f32_16x16x32_f16(false, a1, false, b1,
                                                   (short)0, c, false, false);

        // advance (i,j) to next pair: i < j, row-major over upper triangle
        if (++j == NFIELD) { ++i; j = i + 1; }
    }

    // Spill per-wave 16x16 accumulator to LDS in (vgpr, lane) order.
    float* my = sRed + wave * 256;
#pragma unroll
    for (int r = 0; r < 8; ++r) my[r * 32 + lane] = c[r];
    __syncthreads();

    // Wave 0, lanes 0-15: pull diagonal (sample s): element (s,s) sits at
    // VGPR s&7, lane (s<8 ? s : s+16). Sum the wave partials, add linear
    // term + bias, apply sigmoid, store one f32 per sample.
    if (wave == 0 && lane < 16) {
        const int s  = lane;
        const int da = (s & 7) * 32 + ((s < 8) ? s : (s + 16));
        float inter = 0.f;
#pragma unroll
        for (int w = 0; w < WPB; ++w) inter += sRed[w * 256 + da];

        float lin = bias[0];
#pragma unroll
        for (int f = 0; f < NFIELD; ++f) lin += Wl[sIdx[s * NFIELD + f]];

        const float t = lin + inter;
        out[s0 + s] = 1.0f / (1.0f + __expf(-t));
    }
}

extern "C" void kernel_launch(void* const* d_in, const int* in_sizes, int n_in,
                              void* d_out, int out_size, void* d_ws, size_t ws_size,
                              hipStream_t stream) {
    const int*   x    = (const int*)d_in[0];     // (8192, 20)
    const float* W    = (const float*)d_in[1];   // (20, 100000, 64)
    const float* Wl   = (const float*)d_in[2];   // (100000, 1)
    const float* bias = (const float*)d_in[3];   // (1,)
    float*       out  = (float*)d_out;           // (8192, 1)

    const int tiles = out_size / 16;             // 8192 / 16 = 512 blocks
    ffm_forward<<<tiles, 32 * WPB, 0, stream>>>(x, W, Wl, bias, out);
}
